// SHGR_28243704938825
// MI455X (gfx1250) — matrix operand order repaired and checked by reading, not compiled, not run on hardware
//
#include <hip/hip_runtime.h>
#include <math.h>

// ---------------- types for WMMA (CDNA5 / gfx1250, wave32) ----------------
typedef __bf16 bf16_t;
typedef __bf16 v8bf  __attribute__((ext_vector_type(8)));
typedef __bf16 v16bf __attribute__((ext_vector_type(16)));
typedef float  v8f   __attribute__((ext_vector_type(8)));

#define BQ      2048          // queries
#define NS      100000        // supports
#define DIM     128
#define TDIM    8             // targets per support
#define KNN     16
#define NSPLIT  10            // N split for occupancy: 10 * 10000
#define SEGLEN  (NS / NSPLIT)     // 10000
#define SEGTILES (SEGLEN / 16)    // 625 column tiles of 16
#define WAVES   4
#define CPR     (NSPLIT * WAVES * KNN)   // 640 candidates per query row

// float -> bf16, round-to-nearest-even
static __device__ __forceinline__ unsigned short f2bf(float f) {
  unsigned u = __float_as_uint(f);
  unsigned r = u + 0x7FFFu + ((u >> 16) & 1u);
  return (unsigned short)(r >> 16);
}

// wave-local LDS fence: all lanes' DS writes visible before next read
#define WAVE_LDS_FENCE()                                   \
  do {                                                     \
    __builtin_amdgcn_wave_barrier();                       \
    asm volatile("s_wait_dscnt 0" ::: "memory");           \
    __builtin_amdgcn_wave_barrier();                       \
  } while (0)

// --------------------------------------------------------------------------
// Kernel 1: L2-normalize rows (D=128) and convert to bf16. One wave per row.
// --------------------------------------------------------------------------
__global__ __launch_bounds__(256) void l2norm_bf16(
    const float* __restrict__ x, unsigned short* __restrict__ o, int nrows) {
  const int wv   = threadIdx.x >> 5;
  const int lane = threadIdx.x & 31;
  const int row  = blockIdx.x * 8 + wv;
  if (row >= nrows) return;                      // wave-uniform
  const float4 v = *(const float4*)(x + (size_t)row * DIM + lane * 4);
  float ss = v.x * v.x + v.y * v.y + v.z * v.z + v.w * v.w;
#pragma unroll
  for (int off = 16; off > 0; off >>= 1) ss += __shfl_xor(ss, off, 32);
  const float scale = 1.0f / fmaxf(sqrtf(ss), 1e-12f);
  unsigned lo = (unsigned)f2bf(v.x * scale) | ((unsigned)f2bf(v.y * scale) << 16);
  unsigned hi = (unsigned)f2bf(v.z * scale) | ((unsigned)f2bf(v.w * scale) << 16);
  uint2 r; r.x = lo; r.y = hi;
  *(uint2*)(o + (size_t)row * DIM + lane * 4) = r;
}

// B fragment for one 16-column tile: 4 K-steps of the 32x16 bf16 B layout.
struct BF {
  v16bf x0, x1, x2, x3;
};

// --------------------------------------------------------------------------
// Kernel 2: fused bf16-WMMA similarity GEMM + running per-row top-16.
// B fragments are register double-buffered and loaded UNCONDITIONALLY every
// trip (bounded over-read stays inside the workspace), so the waitcnt pass
// sees a uniform scoreboard and can leave the 8 newer loads in flight
// (s_wait_loadcnt 8) while the current tile's WMMA chain executes.
// --------------------------------------------------------------------------
__global__ __launch_bounds__(128) void knn_topk_wmma(
    const unsigned short* __restrict__ qb_, const unsigned short* __restrict__ sb_,
    float* __restrict__ candd, int* __restrict__ candi) {
  const bf16_t* qb = (const bf16_t*)qb_;
  const bf16_t* sb = (const bf16_t*)sb_;

  __shared__ float tileLds[WAVES][16][16];   // spilled score tile
  __shared__ float simLds [WAVES][16][KNN];  // per-row best sims (unsorted)
  __shared__ int   idxLds [WAVES][16][KNN];
  __shared__ float thLds  [WAVES][16];       // per-row min of best sims

  const int tid  = threadIdx.x;
  const int wv   = tid >> 5;
  const int lane = tid & 31;
  const int half = lane >> 4;   // K-half selector for 16-bit WMMA layouts
  const int m    = lane & 15;   // A row / B column within tile

  const int qt    = blockIdx.x / NSPLIT;
  const int seg   = blockIdx.x % NSPLIT;
  const int qbase = qt * 16;
  const int seg0  = seg * SEGLEN;

  // A fragments (16x32 bf16 per K-step), resident for the whole kernel.
  // Contiguous order, same as B: lane m, half h holds K = 16h+{0..15}.
  v16bf a[4];
  {
    const bf16_t* qrow = qb + (size_t)(qbase + m) * DIM;
#pragma unroll
    for (int k0 = 0; k0 < 4; ++k0) {
      v8bf lo = *(const v8bf*)(qrow + k0 * 32 + 16 * half);
      v8bf hi = *(const v8bf*)(qrow + k0 * 32 + 16 * half + 8);
      a[k0] = __builtin_shufflevector(lo, hi, 0, 1, 2, 3, 4, 5, 6, 7, 8, 9, 10,
                                      11, 12, 13, 14, 15);
    }
  }

  float th[8];                       // thresholds for this lane's 8 C rows
#pragma unroll
  for (int v = 0; v < 8; ++v) th[v] = -1e30f;
  float curmin = -1e30f;             // owner-lane (lane<16) running min
  int   minpos = 0;
  if (lane < 16) {
#pragma unroll
    for (int j = 0; j < KNN; ++j) { simLds[wv][lane][j] = -1e30f; idxLds[wv][lane][j] = 0; }
  }

  // B fragment source: supports row-major [n][k] == K^T-major, lane m = col,
  // half h holds K = 16h+{0..15} per K-step (32B contiguous per lane).
  // NOTE: called with t up to SEGTILES+2*WAVES-1; the bounded over-read
  // (< 64 KB past the bf16 supports) stays inside the workspace allocation.
  auto loadB = [&](int t) -> BF {
    const bf16_t* srow = sb + (size_t)(seg0 + t * 16 + m) * DIM + 16 * half;
    // keep the streamed supports matrix resident in GL2 for other q-tiles
    __builtin_prefetch(srow + (size_t)(2 * WAVES) * 16 * DIM, 0, 3);
    BF r;
    r.x0 = *(const v16bf*)(srow);
    r.x1 = *(const v16bf*)(srow + 32);
    r.x2 = *(const v16bf*)(srow + 64);
    r.x3 = *(const v16bf*)(srow + 96);
    return r;
  };

  // compute one 16x16 score tile, filter vs thresholds, rare top-k insert
  auto tileStep = [&](int t, const BF& B) {
    const int n0 = seg0 + t * 16;
    v8f acc = {0.f, 0.f, 0.f, 0.f, 0.f, 0.f, 0.f, 0.f};
    acc = __builtin_amdgcn_wmma_f32_16x16x32_bf16(false, a[0], false, B.x0,
                                                  (short)0, acc, false, false);
    acc = __builtin_amdgcn_wmma_f32_16x16x32_bf16(false, a[1], false, B.x1,
                                                  (short)0, acc, false, false);
    acc = __builtin_amdgcn_wmma_f32_16x16x32_bf16(false, a[2], false, B.x2,
                                                  (short)0, acc, false, false);
    acc = __builtin_amdgcn_wmma_f32_16x16x32_bf16(false, a[3], false, B.x3,
                                                  (short)0, acc, false, false);

    bool pass = false;
#pragma unroll
    for (int v = 0; v < 8; ++v) pass |= (acc[v] > th[v]);

    if (__builtin_amdgcn_ballot_w32(pass)) {
      // spill C tile: lane holds rows v+8*half, column m
#pragma unroll
      for (int v = 0; v < 8; ++v) tileLds[wv][v + 8 * half][m] = acc[v];
      WAVE_LDS_FENCE();
      if (lane < 16) {               // one owner lane per query row
        const int row = lane;
        float cm = curmin; int mp = minpos;
        for (int j = 0; j < 16; ++j) {
          float sv = tileLds[wv][row][j];
          if (sv > cm) {             // replace current min, rescan for new min
            simLds[wv][row][mp] = sv;
            idxLds[wv][row][mp] = n0 + j;
            cm = 1e30f;
#pragma unroll
            for (int q2 = 0; q2 < KNN; ++q2) {
              float sq = simLds[wv][row][q2];
              if (sq < cm) { cm = sq; mp = q2; }
            }
          }
        }
        curmin = cm; minpos = mp;
        thLds[wv][row] = cm;
      }
      WAVE_LDS_FENCE();
#pragma unroll
      for (int v = 0; v < 8; ++v) th[v] = thLds[wv][v + 8 * half];
    }
  };

  // software-pipelined main loop: ping-pong B buffers, 2 tiles per trip.
  // Loads are unconditional (uniform scoreboard state => non-zero loadcnt
  // waits); compute steps stay guarded by wave-uniform bounds checks, so
  // EXEC is all-ones at every WMMA.
  BF buf0 = loadB(wv);
  BF buf1;
  for (int t = wv; t < SEGTILES; t += 2 * WAVES) {
    buf1 = loadB(t + WAVES);               // issue early, no wait yet
    tileStep(t, buf0);
    buf0 = loadB(t + 2 * WAVES);           // issue early, no wait yet
    if (t + WAVES < SEGTILES) tileStep(t + WAVES, buf1);
  }

  // write this wave's 16 candidates per row as (distance, index)
  if (lane < 16) {
    const int row = lane;
    const size_t base = (((size_t)(qbase + row) * NSPLIT + seg) * WAVES + wv) * KNN;
#pragma unroll
    for (int j = 0; j < KNN; ++j) {
      candd[base + j] = 1.0f - simLds[wv][row][j];
      candi[base + j] = idxLds[wv][row][j];
    }
  }
}

// --------------------------------------------------------------------------
// Kernel 3: per query row, reduce 640 candidates -> top-16, inverse-distance
// weights, gather targets. One thread per row.
// --------------------------------------------------------------------------
__global__ __launch_bounds__(256) void knn_merge(
    const float* __restrict__ candd, const int* __restrict__ candi,
    const float* __restrict__ targets, float* __restrict__ out) {
  __shared__ float bd[256][KNN];
  __shared__ int   bi[256][KNN];
  const int t = threadIdx.x;
  const int g = blockIdx.x * 256 + t;

  float cmax = 1e30f; int mp = 0;
#pragma unroll
  for (int j = 0; j < KNN; ++j) { bd[t][j] = 1e30f; bi[t][j] = 0; }

  const float* cd = candd + (size_t)g * CPR;
  const int*   ci = candi + (size_t)g * CPR;
  for (int c = 0; c < CPR; ++c) {
    float d = cd[c];
    if (d < cmax) {                  // replace current max, rescan
      bd[t][mp] = d; bi[t][mp] = ci[c];
      cmax = -1e30f;
#pragma unroll
      for (int j = 0; j < KNN; ++j) {
        float dj = bd[t][j];
        if (dj > cmax) { cmax = dj; mp = j; }
      }
    }
  }

  float w[KNN]; float wsum = 0.f;
#pragma unroll
  for (int j = 0; j < KNN; ++j) { w[j] = 1.0f / (bd[t][j] + 1e-8f); wsum += w[j]; }
  const float inv = 1.0f / wsum;
#pragma unroll
  for (int tt = 0; tt < TDIM; ++tt) {
    float acc = 0.f;
#pragma unroll
    for (int j = 0; j < KNN; ++j) acc += w[j] * targets[(size_t)bi[t][j] * TDIM + tt];
    out[(size_t)g * TDIM + tt] = acc * inv;
  }
}

// --------------------------------------------------------------------------
extern "C" void kernel_launch(void* const* d_in, const int* in_sizes, int n_in,
                              void* d_out, int out_size, void* d_ws, size_t ws_size,
                              hipStream_t stream) {
  (void)in_sizes; (void)n_in; (void)out_size; (void)ws_size;
  const float* q  = (const float*)d_in[0];   // [2048,128]
  const float* s  = (const float*)d_in[1];   // [100000,128]
  const float* tg = (const float*)d_in[2];   // [100000,8]
  // d_in[3] = k (device scalar) == 16, baked in as KNN

  char* ws = (char*)d_ws;
  unsigned short* sb   = (unsigned short*)(ws);                               // 25.6 MB
  unsigned short* qbuf = (unsigned short*)(ws + (size_t)NS * DIM * 2);        // 0.5 MB
  float* candd = (float*)(ws + (size_t)NS * DIM * 2 + (size_t)BQ * DIM * 2);  // 5.24 MB
  int*   candi = (int*)(ws + (size_t)NS * DIM * 2 + (size_t)BQ * DIM * 2 +
                        (size_t)BQ * CPR * 4);                                // 5.24 MB

  l2norm_bf16<<<NS / 8, 256, 0, stream>>>(s, sb, NS);
  l2norm_bf16<<<BQ / 8, 256, 0, stream>>>(q, qbuf, BQ);
  knn_topk_wmma<<<(BQ / 16) * NSPLIT, 128, 0, stream>>>(qbuf, sb, candd, candi);
  knn_merge<<<BQ / 256, 256, 0, stream>>>(candd, candi, tg, (float*)d_out);
}
